// chebConv_70763881168941
// MI455X (gfx1250) — compile-verified, matrix operand order checked by reading, not run on hardware
//
#include <hip/hip_runtime.h>

#define NROWS 8192
#define CDIM  128
#define BM    64
#define BK    32
#define NT    128   // 4 waves of 32

typedef __attribute__((ext_vector_type(16))) __bf16 v16bf;
typedef __attribute__((ext_vector_type(4)))  __bf16 v4bf;
typedef __attribute__((ext_vector_type(8)))  float  v8f;

union V16U { unsigned int u[8]; v16bf v; };

// One wave computes a 16(M) x 128(N) strip: 8 WMMA tiles, K-chunk of 32.
// A layout (16-bit A 16x32, ISA 7.12.2): lane m=lane&15; VGPR v holds K pair
//   kb = (v<4 ? 2v : 16+2(v-4)) + (lane>=16 ? 8 : 0)  -> one b32 LDS read.
// B kept N-major in LDS (Bs[n][k]) so the same pair is contiguous.
__device__ __forceinline__ void wave_mma(const __bf16 (*As)[BK],
                                         const __bf16 (*Bs)[BK],
                                         int w, int lane, v8f acc[8]) {
  const int m  = lane & 15;
  const int kh = (lane >> 4) * 8;
  V16U a;
#pragma unroll
  for (int v = 0; v < 8; ++v) {
    int kb = (v < 4 ? v * 2 : 16 + (v - 4) * 2) + kh;
    a.u[v] = *(const unsigned int*)&As[w * 16 + m][kb];
  }
#pragma unroll
  for (int t = 0; t < 8; ++t) {
    V16U b;
#pragma unroll
    for (int v = 0; v < 8; ++v) {
      int kb = (v < 4 ? v * 2 : 16 + (v - 4) * 2) + kh;
      b.u[v] = *(const unsigned int*)&Bs[t * 16 + m][kb];
    }
    acc[t] = __builtin_amdgcn_wmma_f32_16x16x32_bf16(
        false, a.v, false, b.v, (short)0, acc[t], false, false);
  }
}

// One-shot convert + transpose: S f32 [8192][128] -> St bf16 [128][8192].
// S is L2-resident (4 MB); output feeds the stage kernels' B path.
__global__ void __launch_bounds__(256)
cvt_transpose(const float* __restrict__ S, __bf16* __restrict__ St) {
  int g  = blockIdx.x * 256 + threadIdx.x;   // 128K threads, 8 k-elems each
  int n  = g >> 10;                          // 0..127
  int k0 = (g & 1023) << 3;                  // 0..8184 step 8
  float a[8];
#pragma unroll
  for (int j = 0; j < 8; ++j) a[j] = S[(size_t)(k0 + j) * CDIM + n];
  v4bf q0 = { (__bf16)a[0], (__bf16)a[1], (__bf16)a[2], (__bf16)a[3] };
  v4bf q1 = { (__bf16)a[4], (__bf16)a[5], (__bf16)a[6], (__bf16)a[7] };
  *(v4bf*)(St + (size_t)n * NROWS + k0)     = q0;   // coalesced b64 stores
  *(v4bf*)(St + (size_t)n * NROWS + k0 + 4) = q1;
}

// D = L @ S            (mode 0, Subp unused)
// D = 2*(L @ S) - Subp (mode 1)
// Stp = S^T pre-converted to bf16, [CDIM][NROWS] n-major.
__global__ void __launch_bounds__(NT, 1)
cheb_stage(const float* __restrict__ Lp, const __bf16* __restrict__ Stp,
           const float* __restrict__ Subp, float* __restrict__ Dp, int mode) {
  __shared__ __bf16 As[2][BM][BK];    // 2 x 4 KB, double-buffered
  __shared__ __bf16 Bs[2][CDIM][BK];  // 2 x 8 KB, double-buffered

  const int tid  = threadIdx.x;
  const int lane = tid & 31;
  const int w    = tid >> 5;
  const int rowBase = blockIdx.x * BM;

  v8f acc[8];
#pragma unroll
  for (int t = 0; t < 8; ++t)
    acc[t] = (v8f){0.f, 0.f, 0.f, 0.f, 0.f, 0.f, 0.f, 0.f};

  float4 ra[4];   // A staging only (f32, needs convert); B never touches VGPRs

  auto gloadA = [&](int k0) {
#pragma unroll
    for (int i = 0; i < 4; ++i) {          // A: 64x32 f32 from L
      int f = tid + i * NT, r = f >> 3, c4 = (f & 7) << 2;
      ra[i] = *(const float4*)(Lp + (size_t)(rowBase + r) * NROWS + k0 + c4);
    }
  };
  // B: 128x32 bf16 tile copied memory -> LDS by the async engine (ASYNCcnt),
  // zero VGPR staging, overlaps the WMMA block.
  auto asyncB = [&](int k0, int buf) {
#pragma unroll
    for (int i = 0; i < 4; ++i) {
      int f = tid + i * NT, n = f >> 2, c8 = (f & 3) << 3;
      const __bf16* g = Stp + (size_t)n * NROWS + k0 + c8;
      unsigned lds = (unsigned)(size_t)&Bs[buf][n][c8];
      asm volatile("global_load_async_to_lds_b128 %0, %1, off"
                   :: "v"(lds), "v"(g) : "memory");
    }
  };
  auto lstoreA = [&](int buf) {
#pragma unroll
    for (int i = 0; i < 4; ++i) {          // native cvt -> one b64 LDS store
      int f = tid + i * NT, r = f >> 3, c4 = (f & 7) << 2;
      v4bf q = { (__bf16)ra[i].x, (__bf16)ra[i].y, (__bf16)ra[i].z, (__bf16)ra[i].w };
      *(v4bf*)&As[buf][r][c4] = q;
    }
  };

  gloadA(0);
  asyncB(0, 0);
  lstoreA(0);
  asm volatile("s_wait_asynccnt 0x0" ::: "memory");
  __syncthreads();

  for (int k0 = 0; k0 < NROWS; k0 += 2 * BK) {
#pragma unroll
    for (int p = 0; p < 2; ++p) {          // p is constant after unroll:
      const int  kc  = k0 + p * BK;        //  read buf p, fill buf 1-p
      const int  kn  = kc + BK;
      const bool nxt = kn < NROWS;
      if (nxt) {
        gloadA(kn);                        // A global loads issued first
        asyncB(kn, 1 - p);                 // B streams straight into LDS
        if (kn + BK < NROWS)               // prefetch L one chunk further
          __builtin_prefetch(Lp + (size_t)(rowBase + (tid >> 3)) * NROWS + kn + BK, 0, 3);
      }
      wave_mma(As[p], Bs[p], w, lane, acc);  // overlaps load latency
      if (nxt) lstoreA(1 - p);             // waits A loads, fills other buffer
      asm volatile("s_wait_asynccnt 0x0" ::: "memory");  // B tile landed
      __syncthreads();                     // single barrier per k-step
    }
  }

  // C/D layout: lane m = N, VGPR e -> M = e (+8 for lanes 16..31)
  const int m = lane & 15, rofs = (lane >> 4) * 8;
#pragma unroll
  for (int t = 0; t < 8; ++t) {
#pragma unroll
    for (int e = 0; e < 8; ++e) {
      int row = rowBase + w * 16 + rofs + e;
      size_t idx = (size_t)row * CDIM + t * 16 + m;
      float vo = acc[t][e];
      if (mode) vo = 2.0f * vo - Subp[idx];
      Dp[idx] = vo;
    }
  }
}

// out = T0@W0^T + T1@W1^T + T2@W2^T  (stacked-K GEMM, Ktot = 384)
__global__ void __launch_bounds__(NT)
cheb_out(const float* __restrict__ T0p, const float* __restrict__ T1p,
         const float* __restrict__ T2p, const float* __restrict__ Wp,
         float* __restrict__ Op) {
  __shared__ __bf16 As[BM][BK];
  __shared__ __bf16 Bs[CDIM][BK];

  const int tid  = threadIdx.x;
  const int lane = tid & 31;
  const int w    = tid >> 5;
  const int rowBase = blockIdx.x * BM;

  v8f acc[8];
#pragma unroll
  for (int t = 0; t < 8; ++t)
    acc[t] = (v8f){0.f, 0.f, 0.f, 0.f, 0.f, 0.f, 0.f, 0.f};

  for (int ks = 0; ks < 12; ++ks) {        // 12 chunks of 32 over Ktot=384
    const int j   = ks >> 2;
    const int cib = (ks & 3) * BK;
    const float* Tj = (j == 0) ? T0p : (j == 1) ? T1p : T2p;

#pragma unroll
    for (int i = 0; i < 4; ++i) {          // A: 64x32 slice of Tj
      int f = tid + i * NT, r = f >> 3, c4 = (f & 7) << 2;
      float4 v = *(const float4*)(Tj + (size_t)(rowBase + r) * CDIM + cib + c4);
      v4bf q = { (__bf16)v.x, (__bf16)v.y, (__bf16)v.z, (__bf16)v.w };
      *(v4bf*)&As[r][c4] = q;
    }
#pragma unroll
    for (int i = 0; i < 8; ++i) {          // B[n][k] = W[j][n][cib+k] (already N-major)
      int f = tid + i * NT, n = f >> 3, c4 = (f & 7) << 2;
      float4 v = *(const float4*)(Wp + (size_t)j * CDIM * CDIM + (size_t)n * CDIM + cib + c4);
      v4bf q = { (__bf16)v.x, (__bf16)v.y, (__bf16)v.z, (__bf16)v.w };
      *(v4bf*)&Bs[n][c4] = q;
    }
    __syncthreads();
    wave_mma(As, Bs, w, lane, acc);
    __syncthreads();
  }

  const int m = lane & 15, rofs = (lane >> 4) * 8;
#pragma unroll
  for (int t = 0; t < 8; ++t) {
#pragma unroll
    for (int e = 0; e < 8; ++e) {
      int row = rowBase + w * 16 + rofs + e;
      Op[(size_t)row * CDIM + t * 16 + m] = acc[t][e];
    }
  }
}

extern "C" void kernel_launch(void* const* d_in, const int* in_sizes, int n_in,
                              void* d_out, int out_size, void* d_ws, size_t ws_size,
                              hipStream_t stream) {
  const float* x = (const float*)d_in[0];  // [8192,128]
  const float* L = (const float*)d_in[1];  // [8192,8192]
  const float* W = (const float*)d_in[2];  // [3,128,128]
  float* out = (float*)d_out;              // [8192,128]

  const size_t NM = (size_t)NROWS * CDIM;  // 1M elements
  float*  T1  = (float*)d_ws;              // 4 MB
  float*  T2  = T1 + NM;                   // 4 MB
  __bf16* xt  = (__bf16*)(T2 + NM);        // 2 MB  (x^T  bf16)
  __bf16* T1t = xt + NM;                   // 2 MB  (T1^T bf16)

  dim3 blk(NT), grd(NROWS / BM);           // 128 blocks x 4 waves
  dim3 cblk(256), cgrd(NM / (256 * 8));    // 512 blocks

  cvt_transpose<<<cgrd, cblk, 0, stream>>>(x, xt);
  cheb_stage<<<grd, blk, 0, stream>>>(L, xt,  nullptr, T1, 0);  // T1 = L@x
  cvt_transpose<<<cgrd, cblk, 0, stream>>>(T1, T1t);
  cheb_stage<<<grd, blk, 0, stream>>>(L, T1t, x,       T2, 1);  // T2 = 2(L@T1) - x
  cheb_out <<<grd, blk, 0, stream>>>(x, T1, T2, W, out);
}
